// LoRAModulatedHyperformerPairBlock_23218593202745
// MI455X (gfx1250) — matrix operand order repaired and checked by reading, not compiled, not run on hardware
//
#include <hip/hip_runtime.h>
#include <hip/hip_bf16.h>

typedef __attribute__((ext_vector_type(16))) _Float16 v16h;
typedef __attribute__((ext_vector_type(8)))  _Float16 v8h;
typedef __attribute__((ext_vector_type(8)))  float    v8f;

#define A_  512
#define F_  128
#define C_  16
#define M_  256
#define R_  4
#define DT_ 512
#define EPSF 1e-6f

// Per-wave LDS fence: all LDS traffic in the fused kernel is wave-private,
// so a DScnt wait is sufficient (no block-wide barrier needed).
__device__ __forceinline__ void wave_lds_fence() {
    asm volatile("s_wait_dscnt 0x0" ::: "memory");
}
__device__ __forceinline__ void wave_async_fence() {
    asm volatile("s_wait_asynccnt 0x0" ::: "memory");
}

// ---------------------------------------------------------------------------
// Prep 1: out[i] = sum_m mod[m] * Wa[m*ncols + i]   (hypernet mat-vec)
// ---------------------------------------------------------------------------
__global__ void k_matvec(const float* __restrict__ mod, const float* __restrict__ Wa,
                         int ncols, float* __restrict__ out) {
    int i = blockIdx.x * blockDim.x + threadIdx.x;
    if (i >= ncols) return;
    float s = 0.f;
    for (int m = 0; m < M_; ++m) s += mod[m] * Wa[m * ncols + i];
    out[i] = s;
}

// ---------------------------------------------------------------------------
// Prep 2: WT[j*din+i] = f16( W[i*dout+j] + sum_r Am[i*R+r]*Bm[r*dout+j] )
// (transposed so WMMA B-fragments are contiguous 32B loads)
// ---------------------------------------------------------------------------
__global__ void k_weffT(const float* __restrict__ W, const float* __restrict__ Am,
                        const float* __restrict__ Bm, int din, int dout,
                        _Float16* __restrict__ WT) {
    int t = blockIdx.x * blockDim.x + threadIdx.x;
    if (t >= din * dout) return;
    int i = t % din, j = t / din;
    float s = W[i * dout + j];
#pragma unroll
    for (int r = 0; r < R_; ++r) s += Am[i * R_ + r] * Bm[r * dout + j];
    WT[j * din + i] = (_Float16)s;
}

// ---------------------------------------------------------------------------
// Prep 3: node layernorm -> act [512,128] f32
// ---------------------------------------------------------------------------
__global__ void k_ln_nodes(const float* __restrict__ x, const float* __restrict__ g,
                           const float* __restrict__ b, float* __restrict__ act) {
    __shared__ float red[128];
    int row = blockIdx.x, tid = threadIdx.x;
    float v = x[row * F_ + tid];
    red[tid] = v; __syncthreads();
    for (int s2 = 64; s2 > 0; s2 >>= 1) { if (tid < s2) red[tid] += red[tid + s2]; __syncthreads(); }
    float mean = red[0] * (1.f / F_);
    __syncthreads();
    float d = v - mean;
    red[tid] = d * d; __syncthreads();
    for (int s2 = 64; s2 > 0; s2 >>= 1) { if (tid < s2) red[tid] += red[tid + s2]; __syncthreads(); }
    float var = red[0] * (1.f / F_);
    act[row * F_ + tid] = d * rsqrtf(var + EPSF) * g[tid] + b[tid];
}

// ---------------------------------------------------------------------------
// Prep 4: left/right [512,16] = (act @ Weff + bias) * node_mask
// ---------------------------------------------------------------------------
__global__ void k_leftright(const float* __restrict__ act,
                            const _Float16* __restrict__ WLT, const float* __restrict__ bL,
                            const _Float16* __restrict__ WRT, const float* __restrict__ bR,
                            const float* __restrict__ mask,
                            float* __restrict__ left, float* __restrict__ right) {
    int a = blockIdx.x;
    int c = threadIdx.x & 15;
    int side = threadIdx.x >> 4;
    const _Float16* WT = side ? WRT : WLT;
    float s = side ? bR[c] : bL[c];
    for (int f = 0; f < F_; ++f) s += act[a * F_ + f] * (float)WT[c * F_ + f];
    s *= mask[a];
    (side ? right : left)[a * C_ + c] = s;
}

// ---------------------------------------------------------------------------
// Main fused kernel: per-wave 16-row tile of the [A*A] row space.
//   edge1 = edge_vec + (left⊗right) @ WeffO + bO
//   act2  = LN(edge1);  h = relu(act2 @ WeffT1 + b1)
//   out   = edge1 + h @ WeffT2 + b2
// 4 waves / block, 64 rows / block, 4096 blocks, 320 wmma per wave-tile.
// edge_vec tile arrives via global_load_async_to_lds (ASYNCcnt), overlapped
// with the outer-product fragment build and all of GEMM1.
// ---------------------------------------------------------------------------
#define WAVES 4
__global__ __launch_bounds__(128, 1)
void k_pair_fused(const float* __restrict__ edge_vec,
                  const float* __restrict__ leftbuf, const float* __restrict__ rightbuf,
                  const _Float16* __restrict__ WO_T, const _Float16* __restrict__ WT1_T,
                  const _Float16* __restrict__ WT2_T,
                  const float* __restrict__ b_outp, const float* __restrict__ b_t1,
                  const float* __restrict__ b_t2,
                  const float* __restrict__ ln2g, const float* __restrict__ ln2b,
                  float* __restrict__ out) {
    __shared__ float    s_edge1[WAVES][16][128];            // 32 KB (async landing + edge1)
    __shared__ _Float16 s_act2 [WAVES][16][128];            // 16 KB
    __shared__ _Float16 s_h    [WAVES][16][32];             //  4 KB
    __shared__ _Float16 s_lrh  [WAVES][16 + 256];           // ~2.2 KB (left row + right tile, f16)
    __shared__ float    s_red  [WAVES][64];                 //  1 KB  (layernorm partials)

    const int wid  = threadIdx.x >> 5;
    const int lane = threadIdx.x & 31;
    const int row  = lane & 15;         // A-fragment row / C-fragment col
    const int half = lane >> 4;
    const int col  = lane & 15;
    const int rowb = half * 8;          // C-fragment row base

    const int r0 = blockIdx.x * 64 + wid * 16;   // 16 consecutive (a,b) rows, same a
    const int a  = r0 >> 9;
    const int b0 = r0 & 511;

    // ---- async copy of the contiguous 8KB edge_vec tile into s_edge1 ----
    // dsaddr = VDST_vgpr + INST_OFFSET ; gaddr = VADDR_vgpr + INST_OFFSET
    {
        unsigned laddr = (unsigned)(unsigned long long)(const void*)&s_edge1[wid][0][0]
                         + (unsigned)(lane * 16);
        const char* gaddr = (const char*)(edge_vec + (size_t)r0 * F_) + lane * 16;
#pragma unroll
        for (int i = 0; i < 16; ++i) {
            asm volatile("global_load_async_to_lds_b128 %0, %1, off offset:%2"
                         :: "v"(laddr), "v"(gaddr), "i"(i * 512) : "memory");
        }
    }

    // Warm L2 for the streamed effective weights (global_prefetch_b8)
    __builtin_prefetch(WO_T  + (threadIdx.x << 6), 0, 1);
    __builtin_prefetch(WT1_T + (threadIdx.x << 6), 0, 1);
    __builtin_prefetch(WT2_T + (threadIdx.x << 6), 0, 1);

    // ---- stage left row [16] and right tile [16][16] in f16 ----
    if (lane < 16) s_lrh[wid][lane] = (_Float16)leftbuf[a * C_ + lane];
#pragma unroll
    for (int g = 0; g < 8; ++g) {
        int idx = lane * 8 + g;                 // 0..255
        s_lrh[wid][16 + idx] = (_Float16)rightbuf[(b0 + (idx >> 4)) * C_ + (idx & 15)];
    }
    wave_lds_fence();

    const _Float16* lrow  = &s_lrh[wid][0];
    const _Float16* rtile = &s_lrh[wid][16];

    // ---- build A-fragments of op tile [16 x 256] in f16 ----
    // A-frag k order used here (contiguous; B loads use the same): lane holds row=lane%16; elems 0..15 -> K=half*16+e
    // within the 32-wide k-block.
    v16h a_op[8];
#pragma unroll
    for (int kb = 0; kb < 8; ++kb) {
#pragma unroll
        for (int e = 0; e < 16; ++e) {
            int k = kb * 32 + half * 16 + e;
            int c = k >> 4, d = k & 15;
            a_op[kb][e] = lrow[c] * rtile[row * 16 + d];
        }
    }

    // ---- GEMM1: edge1 = edge_vec + op @ WeffO + bO  (kept as residual C-frags) ----
    v8f out_acc[8];
    v8f acc1[8];
#pragma unroll
    for (int n = 0; n < 8; ++n) {
        v8f acc = {};
#pragma unroll
        for (int kb = 0; kb < 8; ++kb) {
            v16h bf = *(const v16h*)(WO_T + (n * 16 + col) * 256 + kb * 32 + half * 16);
            acc = __builtin_amdgcn_wmma_f32_16x16x32_f16(false, a_op[kb], false, bf,
                                                         (short)0, acc, false, false);
        }
        acc1[n] = acc;
    }

    wave_async_fence();   // edge tile resident in s_edge1
#pragma unroll
    for (int n = 0; n < 8; ++n) {
        float bias = b_outp[n * 16 + col];
#pragma unroll
        for (int g = 0; g < 8; ++g) {
            float ev = s_edge1[wid][rowb + g][n * 16 + col];
            float v  = acc1[n][g] + ev + bias;
            out_acc[n][g] = v;                               // residual for GEMM3
            s_edge1[wid][rowb + g][n * 16 + col] = v;        // layernorm input
        }
    }
    wave_lds_fence();

    // ---- row layernorm -> act2 f16 (2 lanes per row) ----
    {
        int rr = lane >> 1, part = lane & 1;
        float s = 0.f, ss = 0.f;
        for (int c2 = 0; c2 < 64; ++c2) {
            float x = s_edge1[wid][rr][part * 64 + c2];
            s += x; ss += x * x;
        }
        s_red[wid][lane] = s; s_red[wid][32 + lane] = ss;
        wave_lds_fence();
        float tot  = s_red[wid][rr * 2]      + s_red[wid][rr * 2 + 1];
        float tot2 = s_red[wid][32 + rr * 2] + s_red[wid][32 + rr * 2 + 1];
        float mean = tot * (1.f / F_);
        float var  = tot2 * (1.f / F_) - mean * mean;
        float inv  = rsqrtf(var + EPSF);
        for (int c2 = 0; c2 < 64; ++c2) {
            int cc = part * 64 + c2;
            float x = s_edge1[wid][rr][cc];
            s_act2[wid][rr][cc] = (_Float16)((x - mean) * inv * ln2g[cc] + ln2b[cc]);
        }
    }
    wave_lds_fence();

    // ---- A-fragments of act2 [16 x 128] ----
    v16h a2[4];
#pragma unroll
    for (int kb = 0; kb < 4; ++kb) {
        v8h lo = *(const v8h*)(&s_act2[wid][row][kb * 32 + half * 16]);
        v8h hi = *(const v8h*)(&s_act2[wid][row][kb * 32 + half * 16 + 8]);
#pragma unroll
        for (int e = 0; e < 8; ++e) { a2[kb][e] = lo[e]; a2[kb][8 + e] = hi[e]; }
    }

    // ---- GEMM2 (h = relu(act2@Wt1+b1)) fused with GEMM3 (out += h@Wt2) ----
    // Each t covers a 32-wide K-slab of h; h never leaves LDS.
    for (int t = 0; t < 16; ++t) {
#pragma unroll
        for (int u = 0; u < 2; ++u) {
            int n2 = t * 2 + u;
            v8f hacc = {};
#pragma unroll
            for (int kb = 0; kb < 4; ++kb) {
                v16h bf = *(const v16h*)(WT1_T + (n2 * 16 + col) * 128 + kb * 32 + half * 16);
                hacc = __builtin_amdgcn_wmma_f32_16x16x32_f16(false, a2[kb], false, bf,
                                                              (short)0, hacc, false, false);
            }
            float bias = b_t1[n2 * 16 + col];
#pragma unroll
            for (int g = 0; g < 8; ++g) {
                float v = hacc[g] + bias;
                v = v > 0.f ? v : 0.f;
                s_h[wid][rowb + g][u * 16 + col] = (_Float16)v;
            }
        }
        wave_lds_fence();
        v16h a3;
        {
            v8h lo = *(const v8h*)(&s_h[wid][row][half * 16]);
            v8h hi = *(const v8h*)(&s_h[wid][row][half * 16 + 8]);
#pragma unroll
            for (int e = 0; e < 8; ++e) { a3[e] = lo[e]; a3[8 + e] = hi[e]; }
        }
#pragma unroll
        for (int n3 = 0; n3 < 8; ++n3) {
            v16h bf = *(const v16h*)(WT2_T + (n3 * 16 + col) * 512 + t * 32 + half * 16);
            out_acc[n3] = __builtin_amdgcn_wmma_f32_16x16x32_f16(false, a3, false, bf,
                                                                 (short)0, out_acc[n3], false, false);
        }
        wave_lds_fence();   // protect s_h against next iteration's stores (WAR)
    }

    // ---- store out = edge1 + h@Wt2 + b2 ----
#pragma unroll
    for (int n = 0; n < 8; ++n) {
        float bias = b_t2[n * 16 + col];
#pragma unroll
        for (int g = 0; g < 8; ++g) {
            out[(size_t)(r0 + rowb + g) * F_ + n * 16 + col] = out_acc[n][g] + bias;
        }
    }
}

// ---------------------------------------------------------------------------
// Host launch
// ---------------------------------------------------------------------------
extern "C" void kernel_launch(void* const* d_in, const int* in_sizes, int n_in,
                              void* d_out, int out_size, void* d_ws, size_t ws_size,
                              hipStream_t stream) {
    // Input order (setup_inputs dict, depth-first):
    // 0 node_vec 1 edge_vec 2 node_mask 3 edge_mask 4 modulated_params
    // 5 ln1_g 6 ln1_b 7 ln2_g 8 ln2_b
    // 9..12  left  {W,b,Wa,Wb} ; 13..16 right ; 17..20 outp ; 21..24 t1 ; 25..28 t2
    const float* mod = (const float*)d_in[4];

    // Workspace layout
    float* w = (float*)d_ws;
    float* L_Am  = w + 0;     float* L_Bm  = w + 512;
    float* R_Am  = w + 576;   float* R_Bm  = w + 1088;
    float* O_Am  = w + 1152;  float* O_Bm  = w + 2176;
    float* T1_Am = w + 2688;  float* T1_Bm = w + 3200;
    float* T2_Am = w + 5248;  float* T2_Bm = w + 7296;   // ends at 7808 floats

    _Float16* hb   = (_Float16*)((char*)d_ws + 32768);
    _Float16* WL_T  = hb;            // 16x128
    _Float16* WR_T  = hb + 2048;     // 16x128
    _Float16* WO_T  = hb + 4096;     // 128x256
    _Float16* WT1_T = hb + 36864;    // 512x128
    _Float16* WT2_T = hb + 102400;   // 128x512  (ends 167936 halfs)

    float* act   = (float*)((char*)d_ws + 368640);   // 512x128
    float* leftb = (float*)((char*)d_ws + 630784);   // 512x16
    float* rightb= (float*)((char*)d_ws + 663552);   // 512x16

    // --- hypernet mat-vecs: Am = mod@Wa, Bm = mod@Wb ---
    struct MV { const float* Wa; int n; float* out; };
    MV mv[10] = {
        { (const float*)d_in[11], 128 * R_,  L_Am  }, { (const float*)d_in[12], R_ * 16,  L_Bm  },
        { (const float*)d_in[15], 128 * R_,  R_Am  }, { (const float*)d_in[16], R_ * 16,  R_Bm  },
        { (const float*)d_in[19], 256 * R_,  O_Am  }, { (const float*)d_in[20], R_ * 128, O_Bm  },
        { (const float*)d_in[23], 128 * R_,  T1_Am }, { (const float*)d_in[24], R_ * 512, T1_Bm },
        { (const float*)d_in[27], 512 * R_,  T2_Am }, { (const float*)d_in[28], R_ * 128, T2_Bm },
    };
    for (int i = 0; i < 10; ++i)
        k_matvec<<<(mv[i].n + 255) / 256, 256, 0, stream>>>(mod, mv[i].Wa, mv[i].n, mv[i].out);

    // --- effective weights (transposed, f16) ---
    struct WE { const float* W; const float* Am; const float* Bm; int din, dout; _Float16* WT; };
    WE we[5] = {
        { (const float*)d_in[9],  L_Am,  L_Bm,  128, 16,  WL_T  },
        { (const float*)d_in[13], R_Am,  R_Bm,  128, 16,  WR_T  },
        { (const float*)d_in[17], O_Am,  O_Bm,  256, 128, WO_T  },
        { (const float*)d_in[21], T1_Am, T1_Bm, 128, 512, WT1_T },
        { (const float*)d_in[25], T2_Am, T2_Bm, 512, 128, WT2_T },
    };
    for (int i = 0; i < 5; ++i) {
        int n = we[i].din * we[i].dout;
        k_weffT<<<(n + 255) / 256, 256, 0, stream>>>(we[i].W, we[i].Am, we[i].Bm,
                                                     we[i].din, we[i].dout, we[i].WT);
    }

    // --- node layernorm + left/right projections ---
    k_ln_nodes<<<A_, 128, 0, stream>>>((const float*)d_in[0], (const float*)d_in[5],
                                       (const float*)d_in[6], act);
    k_leftright<<<A_, 32, 0, stream>>>(act, WL_T, (const float*)d_in[10],
                                       WR_T, (const float*)d_in[14],
                                       (const float*)d_in[2], leftb, rightb);

    // --- fused pair block: 262144 rows / 64 per block ---
    k_pair_fused<<<(A_ * A_) / 64, 128, 0, stream>>>(
        (const float*)d_in[1], leftb, rightb,
        WO_T, WT1_T, WT2_T,
        (const float*)d_in[18], (const float*)d_in[22], (const float*)d_in[26],
        (const float*)d_in[7], (const float*)d_in[8],
        (float*)d_out);
}
